// MultiHeadAttention_7851200217195
// MI455X (gfx1250) — matrix operand order repaired and checked by reading, not compiled, not run on hardware
//
#include <hip/hip_runtime.h>
#include <hip/hip_bf16.h>
#include <math.h>

// ---------------------------------------------------------------------------
// MI455X (gfx1250) fused multi-head attention + residual + LayerNorm.
// bf16 WMMA (f32 accumulate) everywhere; TDM stages A-panels into LDS.
// Ping-pong double buffering (no register-copy rotation).
// ---------------------------------------------------------------------------

typedef __bf16 bf16_t;
typedef __attribute__((ext_vector_type(16))) __bf16 bf16x16;
typedef __attribute__((ext_vector_type(8)))  __bf16 bf16x8;
typedef __attribute__((ext_vector_type(8)))  float  v8f;
typedef unsigned int       u32;
typedef unsigned long long u64;
typedef __attribute__((ext_vector_type(4))) u32 u32x4;
typedef __attribute__((ext_vector_type(4))) int i32x4;
typedef __attribute__((ext_vector_type(8))) int i32x8;

#define B_DIM   2
#define S_DIM   2048
#define H_DIM   1024
#define NHEAD   16
#define HDIM    64
#define ROWS    (B_DIM * S_DIM)      // 4096
#define EPS_LN  1e-12f
#define NKT     (S_DIM / 32)         // 64 key tiles

// Padded LDS A-panel: 16 rows x 1024 bf16, 16B pad every 512 elems (256 dwords)
// => row pitch 1040 elems (2080B), element (m,k) at  m*1040 + k + ((k>>9)<<3).
#define APITCH  1040
#define ASHARE_ELEMS (16 * APITCH + 8)

__device__ __forceinline__ int lds_a_off(int m, int k) {
  return m * APITCH + k + ((k >> 9) << 3);
}

// ---------------- WMMA helpers ---------------------------------------------

__device__ __forceinline__ v8f wmma_bf16(bf16x16 a, bf16x16 b, v8f c) {
  return __builtin_amdgcn_wmma_f32_16x16x32_bf16(
      false, a, false, b, (short)0, c, false, false);
}

// A 16x32 fragment from row-major global memory.
__device__ __forceinline__ bf16x16 load_a_frag(const bf16_t* base, int ld, int kbase) {
  const int lane = threadIdx.x & 31;
  const int m = lane & 15, g = lane >> 4;
  const bf16_t* row = base + (size_t)m * ld + kbase + g * 8;
  bf16x8 lo = *(const bf16x8*)(row);
  bf16x8 hi = *(const bf16x8*)(row + 16);
  bf16x16 a;
#pragma unroll
  for (int i = 0; i < 8; ++i) { a[i] = lo[i]; a[i + 8] = hi[i]; }
  return a;
}

// A 16x32 fragment from the padded LDS panel (ds_load path).
__device__ __forceinline__ bf16x16 lds_a_frag(const bf16_t* sh, int kbase) {
  const int lane = threadIdx.x & 31;
  const int m = lane & 15, g = lane >> 4;
  const int k0 = kbase + g * 8;
  const int k1 = kbase + 16 + g * 8;
  bf16x8 lo = *(const bf16x8*)(sh + lds_a_off(m, k0));
  bf16x8 hi = *(const bf16x8*)(sh + lds_a_off(m, k1));
  bf16x16 a;
#pragma unroll
  for (int i = 0; i < 8; ++i) { a[i] = lo[i]; a[i + 8] = hi[i]; }
  return a;
}

// B 32x16 fragment from B^T stored row-major (rows = N, cols = K).
__device__ __forceinline__ bf16x16 load_b_frag(const bf16_t* baseT, int ld, int kbase) {
  const int lane = threadIdx.x & 31;
  const int n = lane & 15, g = lane >> 4;
  const bf16_t* row = baseT + (size_t)n * ld + kbase + g * 8;
  bf16x8 lo = *(const bf16x8*)(row);
  bf16x8 hi = *(const bf16x8*)(row + 16);
  bf16x16 b;
#pragma unroll
  for (int i = 0; i < 8; ++i) { b[i] = lo[i]; b[i + 8] = hi[i]; }
  return b;
}

// ---------------- TDM: stage a 16 x 1024 bf16 panel into LDS ----------------
// D# per CDNA5 ISA ch.8: 2D tile, data_size=2B, pad 4 dwords per 256 dwords.

__device__ __forceinline__ void tdm_load_panel(const bf16_t* gsrc, bf16_t* ldsdst) {
#if __has_builtin(__builtin_amdgcn_tensor_load_to_lds)
  const u64 ga   = (u64)(__UINTPTR_TYPE__)gsrc;
  const u32 lo   = __builtin_amdgcn_readfirstlane((u32)ga);
  const u32 hi   = __builtin_amdgcn_readfirstlane((u32)(ga >> 32));
  const u32 loff = __builtin_amdgcn_readfirstlane((u32)(__UINTPTR_TYPE__)ldsdst);

  u32x4 g0;
  g0[0] = 1u;                                   // count=1, user descriptor
  g0[1] = loff;                                 // lds_addr
  g0[2] = lo;                                   // global_addr[31:0]
  g0[3] = (hi & 0x01FFFFFFu) | (2u << 30);      // global_addr[56:32] | type=2

  i32x8 g1;
  g1[0] = (int)((1u << 16)                      // data_size = 2 bytes
              | (1u << 20)                      // pad_enable
              | (7u << 22)                      // pad_interval code 7 = 256 dwords
              | (3u << 25));                    // pad_amount  code 3 = 4 dwords
  g1[1] = (int)(((u32)H_DIM & 0xFFFFu) << 16);  // tensor_dim0 lo16 (bits 63:48)
  g1[2] = (int)(((u32)H_DIM >> 16) | (16u << 16)); // dim0 hi | tensor_dim1 lo16
  g1[3] = (int)(((u32)H_DIM) << 16);            // dim1 hi (0) | tile_dim0 = 1024
  g1[4] = (int)16u;                             // tile_dim1 = 16, tile_dim2 = 0
  g1[5] = (int)H_DIM;                           // tensor_dim0_stride lo32
  g1[6] = 0;
  g1[7] = 0;

  i32x4 z4 = {0, 0, 0, 0};
#if __clang_major__ >= 23
  i32x8 z8 = {0, 0, 0, 0, 0, 0, 0, 0};
  __builtin_amdgcn_tensor_load_to_lds(g0, g1, z4, z4, z8, 0);
#else
  __builtin_amdgcn_tensor_load_to_lds(g0, g1, z4, z4, 0);
#endif
  __builtin_amdgcn_s_wait_tensorcnt(0);
#else
  for (int i = threadIdx.x & 31; i < 16 * H_DIM; i += 32) {
    int m = i >> 10, k = i & (H_DIM - 1);
    ldsdst[lds_a_off(m, k)] = gsrc[(size_t)m * H_DIM + k];
  }
#endif
}

// ---------------- Kernel 1a: f32 -> bf16 elementwise ------------------------

__global__ void cvt_bf16_kernel(const float* __restrict__ in,
                                bf16_t* __restrict__ out, int count) {
  int idx = blockIdx.x * blockDim.x + threadIdx.x;
  if (idx < count) out[idx] = (bf16_t)in[idx];
}

// ---------------- Kernel 1b: f32 1024x1024 -> bf16 transposed ---------------

__global__ void cvt_transpose_kernel(const float* __restrict__ in,
                                     bf16_t* __restrict__ out) {
  __shared__ float tile[32][33];
  const int bx = blockIdx.x * 32, by = blockIdx.y * 32;
  const int tx = threadIdx.x, ty = threadIdx.y;   // blockDim (32,8)
#pragma unroll
  for (int i = 0; i < 32; i += 8)
    tile[ty + i][tx] = in[(size_t)(by + ty + i) * H_DIM + bx + tx];
  __syncthreads();
#pragma unroll
  for (int i = 0; i < 32; i += 8)
    out[(size_t)(bx + ty + i) * H_DIM + by + tx] = (bf16_t)tile[tx][ty + i];
}

// ---------------- Kernel 2: fused QKV projection ----------------------------
// Block = 512 threads (16 waves) = one (mat, 16-row tile).  TDM stages the
// shared X panel into LDS; wave w computes a 16x64 tile.  K-loop is unrolled
// by 2 with ping-pong B-fragment buffers (no rotation copies).

__global__ void
__launch_bounds__(512)
qkv_kernel(const bf16_t* __restrict__ Xbf, const bf16_t* __restrict__ WT,
           const float* __restrict__ bq, const float* __restrict__ bk,
           const float* __restrict__ bv,
           bf16_t* __restrict__ Qbf, bf16_t* __restrict__ Kbf,
           bf16_t* __restrict__ Vt) {
  __shared__ __align__(16) bf16_t ashare[ASHARE_ELEMS];

  const int mat = blockIdx.x >> 8;
  const int rt  = blockIdx.x & 255;
  const int w   = threadIdx.x >> 5;
  const int lane = threadIdx.x & 31;
  const int lm = lane & 15, g = lane >> 4;

  if (threadIdx.x < 32)
    tdm_load_panel(Xbf + (size_t)rt * 16 * H_DIM, ashare);
  __syncthreads();

  const bf16_t* wt = WT + (size_t)mat * (H_DIM * H_DIM)
                        + (size_t)(w * 64) * H_DIM;

  v8f acc[4] = {};
  bf16x16 bA[4], bB[4];
#pragma unroll
  for (int j = 0; j < 4; ++j)
    bA[j] = load_b_frag(wt + (size_t)(j * 16) * H_DIM, H_DIM, 0);

  for (int kb = 0; kb < H_DIM; kb += 64) {
    // half 1: consume bA (kb), prefetch bB (kb+32)
#pragma unroll
    for (int j = 0; j < 4; ++j)
      bB[j] = load_b_frag(wt + (size_t)(j * 16) * H_DIM, H_DIM, kb + 32);
    bf16x16 a0 = lds_a_frag(ashare, kb);
#pragma unroll
    for (int j = 0; j < 4; ++j) acc[j] = wmma_bf16(a0, bA[j], acc[j]);

    // half 2: consume bB (kb+32), prefetch bA (kb+64)
    const int kn = (kb + 64 < H_DIM) ? kb + 64 : 0;
#pragma unroll
    for (int j = 0; j < 4; ++j)
      bA[j] = load_b_frag(wt + (size_t)(j * 16) * H_DIM, H_DIM, kn);
    bf16x16 a1 = lds_a_frag(ashare, kb + 32);
#pragma unroll
    for (int j = 0; j < 4; ++j) acc[j] = wmma_bf16(a1, bB[j], acc[j]);
  }

  const float* bias = (mat == 0) ? bq : (mat == 1) ? bk : bv;
#pragma unroll
  for (int j = 0; j < 4; ++j) {
    const int col = w * 64 + j * 16 + lm;
    const float bb = bias[col];
    if (mat < 2) {
      bf16_t* dst = (mat == 0) ? Qbf : Kbf;
#pragma unroll
      for (int r = 0; r < 8; ++r) {
        const int row = rt * 16 + r + 8 * g;
        dst[(size_t)row * H_DIM + col] = (bf16_t)(acc[j][r] + bb);
      }
    } else {
      bf16x8 v;
#pragma unroll
      for (int r = 0; r < 8; ++r) v[r] = (bf16_t)(acc[j][r] + bb);
      *(bf16x8*)(Vt + (size_t)col * ROWS + rt * 16 + 8 * g) = v;  // V^T packed
    }
  }
}

// ---------------- Kernel 3: flash attention ---------------------------------
// One wave per (b, h, 16-row q-tile).  32-key tiles; pipelined: V fragments
// load first, score WMMAs consume pre-loaded K fragments, next tile's K
// fragments load during the softmax VALU stretch.

__device__ __forceinline__ void attn_step(
    int kt, const bf16_t* kpane, const bf16_t* vpane,
    const bf16x16& aq0, const bf16x16& aq1,
    bf16x16 (&bku)[4], bf16x16 (&bkn)[4],
    float (&mrow)[8], float (&lrow)[8], v8f (&cacc)[4],
    float* pbuf, int lm, int g) {
  const int kb = kt * 32;

  bf16x16 bv0 = load_b_frag(vpane + (size_t)0  * ROWS, ROWS, kb);
  bf16x16 bv1 = load_b_frag(vpane + (size_t)16 * ROWS, ROWS, kb);
  bf16x16 bv2 = load_b_frag(vpane + (size_t)32 * ROWS, ROWS, kb);
  bf16x16 bv3 = load_b_frag(vpane + (size_t)48 * ROWS, ROWS, kb);

  v8f s0 = {}, s1 = {};
  s0 = wmma_bf16(aq0, bku[0], s0);
  s0 = wmma_bf16(aq1, bku[1], s0);
  s1 = wmma_bf16(aq0, bku[2], s1);
  s1 = wmma_bf16(aq1, bku[3], s1);

  // prefetch next key tile's K fragments (overlaps softmax below)
  const int ktn = (kt + 1 < NKT) ? kt + 1 : kt;
  const bf16_t* nk0 = kpane + (size_t)(ktn * 32) * H_DIM;
  const bf16_t* nk1 = nk0 + (size_t)16 * H_DIM;
  bkn[0] = load_b_frag(nk0, H_DIM, 0);
  bkn[1] = load_b_frag(nk0, H_DIM, 32);
  bkn[2] = load_b_frag(nk1, H_DIM, 0);
  bkn[3] = load_b_frag(nk1, H_DIM, 32);
  __builtin_prefetch(vpane + kb + 32, 0, 0);   // next V tile -> caches

#pragma unroll
  for (int r = 0; r < 8; ++r) {
    float v0 = s0[r] * 0.125f;                 // 1/sqrt(HD)
    float v1 = s1[r] * 0.125f;
    float mx = fmaxf(v0, v1);
#pragma unroll
    for (int off = 1; off < 16; off <<= 1) mx = fmaxf(mx, __shfl_xor(mx, off, 16));
    const float mn = fmaxf(mrow[r], mx);
    const float sf = __expf(mrow[r] - mn);
    const float p0 = __expf(v0 - mn);
    const float p1 = __expf(v1 - mn);
    float ps = p0 + p1;
#pragma unroll
    for (int off = 1; off < 16; off <<= 1) ps += __shfl_xor(ps, off, 16);
    lrow[r] = lrow[r] * sf + ps;
    mrow[r] = mn;
#pragma unroll
    for (int j = 0; j < 4; ++j) cacc[j][r] *= sf;
    pbuf[(r + 8 * g) * 32 + lm]      = p0;
    pbuf[(r + 8 * g) * 32 + 16 + lm] = p1;
  }

  bf16x16 ap;   // P: C-layout f32 -> A-fragment bf16 via LDS
#pragma unroll
  for (int i = 0; i < 8; ++i) {
    ap[i]     = (bf16_t)pbuf[lm * 32 + g * 8 + i];
    ap[i + 8] = (bf16_t)pbuf[lm * 32 + 16 + g * 8 + i];
  }

  cacc[0] = wmma_bf16(ap, bv0, cacc[0]);
  cacc[1] = wmma_bf16(ap, bv1, cacc[1]);
  cacc[2] = wmma_bf16(ap, bv2, cacc[2]);
  cacc[3] = wmma_bf16(ap, bv3, cacc[3]);
}

__global__ void
__launch_bounds__(32)
attn_kernel(const bf16_t* __restrict__ Qbf, const bf16_t* __restrict__ Kbf,
            const bf16_t* __restrict__ Vt, bf16_t* __restrict__ Ctx) {
  __shared__ float pbuf[16 * 32];

  const int bh = blockIdx.x >> 7;
  const int qt = blockIdx.x & 127;
  const int b  = bh >> 4, h = bh & 15;
  const int lane = threadIdx.x & 31;
  const int lm = lane & 15, g = lane >> 4;
  const int rowbase = b * S_DIM + qt * 16;

  const bf16_t* qrow = Qbf + (size_t)rowbase * H_DIM + h * HDIM;
  bf16x16 aq0 = load_a_frag(qrow, H_DIM, 0);
  bf16x16 aq1 = load_a_frag(qrow, H_DIM, 32);

  const bf16_t* kpane = Kbf + (size_t)(b * S_DIM) * H_DIM + h * HDIM;
  const bf16_t* vpane = Vt + (size_t)(h * HDIM) * ROWS + b * S_DIM;

  float mrow[8], lrow[8];
#pragma unroll
  for (int r = 0; r < 8; ++r) { mrow[r] = -1e30f; lrow[r] = 0.f; }
  v8f cacc[4] = {};

  bf16x16 bkA[4], bkB[4];
  bkA[0] = load_b_frag(kpane, H_DIM, 0);
  bkA[1] = load_b_frag(kpane, H_DIM, 32);
  bkA[2] = load_b_frag(kpane + (size_t)16 * H_DIM, H_DIM, 0);
  bkA[3] = load_b_frag(kpane + (size_t)16 * H_DIM, H_DIM, 32);

  for (int kt = 0; kt < NKT; kt += 2) {
    attn_step(kt,     kpane, vpane, aq0, aq1, bkA, bkB, mrow, lrow, cacc, pbuf, lm, g);
    attn_step(kt + 1, kpane, vpane, aq0, aq1, bkB, bkA, mrow, lrow, cacc, pbuf, lm, g);
  }

#pragma unroll
  for (int j = 0; j < 4; ++j) {
    const int col = h * HDIM + j * 16 + lm;
#pragma unroll
    for (int r = 0; r < 8; ++r) {
      const int row = rowbase + r + 8 * g;
      Ctx[(size_t)row * H_DIM + col] = (bf16_t)(cacc[j][r] / lrow[r]);
    }
  }
}

// ---------------- Kernel 4: output projection + residual + LayerNorm --------

__global__ void
__launch_bounds__(512)
proj_ln_kernel(const bf16_t* __restrict__ Ctx, const bf16_t* __restrict__ WoT,
               const float* __restrict__ bo, const float* __restrict__ X,
               const float* __restrict__ gamma, const float* __restrict__ beta,
               float* __restrict__ out) {
  __shared__ __align__(16) bf16_t ashare[ASHARE_ELEMS];
  __shared__ float psum[16][16];
  __shared__ float psq[16][16];
  __shared__ float stats_mu[16];
  __shared__ float stats_rs[16];

  const int rb = blockIdx.x * 16;
  const int w  = threadIdx.x >> 5;
  const int lane = threadIdx.x & 31;
  const int lm = lane & 15, g = lane >> 4;

  if (threadIdx.x < 32)
    tdm_load_panel(Ctx + (size_t)rb * H_DIM, ashare);
  __syncthreads();

  const bf16_t* wt = WoT + (size_t)(w * 64) * H_DIM;

  v8f acc[4] = {};
  bf16x16 bA[4], bB[4];
#pragma unroll
  for (int j = 0; j < 4; ++j)
    bA[j] = load_b_frag(wt + (size_t)(j * 16) * H_DIM, H_DIM, 0);

  for (int kb = 0; kb < H_DIM; kb += 64) {
#pragma unroll
    for (int j = 0; j < 4; ++j)
      bB[j] = load_b_frag(wt + (size_t)(j * 16) * H_DIM, H_DIM, kb + 32);
    bf16x16 a0 = lds_a_frag(ashare, kb);
#pragma unroll
    for (int j = 0; j < 4; ++j) acc[j] = wmma_bf16(a0, bA[j], acc[j]);

    const int kn = (kb + 64 < H_DIM) ? kb + 64 : 0;
#pragma unroll
    for (int j = 0; j < 4; ++j)
      bA[j] = load_b_frag(wt + (size_t)(j * 16) * H_DIM, H_DIM, kn);
    bf16x16 a1 = lds_a_frag(ashare, kb + 32);
#pragma unroll
    for (int j = 0; j < 4; ++j) acc[j] = wmma_bf16(a1, bB[j], acc[j]);
  }

  // bias + residual in registers, per-row partial sums for LayerNorm
#pragma unroll
  for (int j = 0; j < 4; ++j) {
    const int col = w * 64 + j * 16 + lm;
    const float bb = bo[col];
#pragma unroll
    for (int r = 0; r < 8; ++r) {
      const int row = rb + r + 8 * g;
      acc[j][r] = acc[j][r] + bb + X[(size_t)row * H_DIM + col];
    }
  }
#pragma unroll
  for (int r = 0; r < 8; ++r) {
    float s = 0.f, s2 = 0.f;
#pragma unroll
    for (int j = 0; j < 4; ++j) { float v = acc[j][r]; s += v; s2 += v * v; }
#pragma unroll
    for (int off = 1; off < 16; off <<= 1) {
      s  += __shfl_xor(s,  off, 16);
      s2 += __shfl_xor(s2, off, 16);
    }
    if (lm == 0) { psum[r + 8 * g][w] = s; psq[r + 8 * g][w] = s2; }
  }
  __syncthreads();

  if (threadIdx.x < 16) {
    float s = 0.f, s2 = 0.f;
#pragma unroll
    for (int ww = 0; ww < 16; ++ww) { s += psum[threadIdx.x][ww]; s2 += psq[threadIdx.x][ww]; }
    const float mu  = s * (1.f / H_DIM);
    const float var = s2 * (1.f / H_DIM) - mu * mu;
    stats_mu[threadIdx.x] = mu;
    stats_rs[threadIdx.x] = rsqrtf(var + EPS_LN);
  }
  __syncthreads();

#pragma unroll
  for (int j = 0; j < 4; ++j) {
    const int col = w * 64 + j * 16 + lm;
    const float gm = gamma[col], bt = beta[col];
#pragma unroll
    for (int r = 0; r < 8; ++r) {
      const int rl = r + 8 * g;
      out[(size_t)(rb + rl) * H_DIM + col] =
          (acc[j][r] - stats_mu[rl]) * stats_rs[rl] * gm + bt;
    }
  }
}

// ---------------- Host launch ------------------------------------------------

extern "C" void kernel_launch(void* const* d_in, const int* in_sizes, int n_in,
                              void* d_out, int out_size, void* d_ws, size_t ws_size,
                              hipStream_t stream) {
  const float* hs    = (const float*)d_in[0];
  const float* Wq    = (const float*)d_in[1];
  const float* bq    = (const float*)d_in[2];
  const float* Wk    = (const float*)d_in[3];
  const float* bk    = (const float*)d_in[4];
  const float* Wv    = (const float*)d_in[5];
  const float* bv    = (const float*)d_in[6];
  const float* Wo    = (const float*)d_in[7];
  const float* bo    = (const float*)d_in[8];
  const float* gamma = (const float*)d_in[9];
  const float* beta  = (const float*)d_in[10];
  float* out = (float*)d_out;

  bf16_t* Xbf = (bf16_t*)d_ws;                       // 4M elems
  bf16_t* WqT = Xbf + (size_t)ROWS * H_DIM;          // 1M each
  bf16_t* WkT = WqT + (size_t)H_DIM * H_DIM;
  bf16_t* WvT = WkT + (size_t)H_DIM * H_DIM;
  bf16_t* WoT = WvT + (size_t)H_DIM * H_DIM;
  bf16_t* Qbf = WoT + (size_t)H_DIM * H_DIM;         // 4M
  bf16_t* Kbf = Qbf + (size_t)ROWS * H_DIM;          // 4M
  bf16_t* Vt  = Kbf + (size_t)ROWS * H_DIM;          // 4M ([H][ROWS])
  bf16_t* Ctx = Vt  + (size_t)ROWS * H_DIM;          // 4M

  const int xcount = ROWS * H_DIM;
  cvt_bf16_kernel<<<xcount / 256, 256, 0, stream>>>(hs, Xbf, xcount);
  dim3 tb(32, 8), tg(H_DIM / 32, H_DIM / 32);
  cvt_transpose_kernel<<<tg, tb, 0, stream>>>(Wq, WqT);
  cvt_transpose_kernel<<<tg, tb, 0, stream>>>(Wk, WkT);
  cvt_transpose_kernel<<<tg, tb, 0, stream>>>(Wv, WvT);
  cvt_transpose_kernel<<<tg, tb, 0, stream>>>(Wo, WoT);

  qkv_kernel<<<3 * 256, 512, 0, stream>>>(Xbf, WqT, bq, bk, bv, Qbf, Kbf, Vt);
  attn_kernel<<<B_DIM * NHEAD * (S_DIM / 16), 32, 0, stream>>>(Qbf, Kbf, Vt, Ctx);
  proj_ln_kernel<<<ROWS / 16, 512, 0, stream>>>(Ctx, WoT, bo, hs, gamma, beta, out);
}
